// OutputBlock_7275674599723
// MI455X (gfx1250) — matrix operand order repaired and checked by reading, not compiled, not run on hardware
//
#include <hip/hip_runtime.h>

#define HIDDEN   128
#define NRAD     16
#define NNODES   50000
#define NEDGES   600000

#define WSTRIDE  136   // padded K stride (halves) for K=128 rows (272B: 16B-aligned, breaks bank stride)
#define W1STRIDE 40    // padded K stride (halves) for K=32 (padded from 16) rW1 rows

// f16 weight-image layout in workspace (halves):
//   [0, 5120)               : rW1^T  (128 cols x 40, zeros beyond k=15)
//   [5120, 22528)           : rW2^T  (128 cols x 136)
//   [22528 + l*17408, ...)  : dWl^T  (128 cols x 136), l = 0..2
#define IMG_W1_OFF   0
#define IMG_W2_OFF   5120
#define IMG_K1_HALF  22528        // kernel-1 combined image size (halves)
#define IMG_DENSE(l) (22528 + (l) * 17408)
#define IMG_TOTAL    (22528 + 3 * 17408)

typedef __attribute__((ext_vector_type(16))) _Float16 v16h;
typedef __attribute__((ext_vector_type(8)))  _Float16 h8;
typedef __attribute__((ext_vector_type(8)))  float    v8f;
typedef __attribute__((ext_vector_type(4)))  float    f4;
typedef __attribute__((ext_vector_type(4)))  unsigned int u32x4;
typedef __attribute__((ext_vector_type(8)))  int      i32x8;
typedef __attribute__((ext_vector_type(4)))  int      i32x4;

__device__ __forceinline__ float silu_f(float v) {
    return v / (1.0f + __expf(-v));
}

// B fragment: two 8-half runs (col-major weight row), two 16B LDS loads.
// Layout: lane&15 = column, halves 0..7 -> K = 8*(lane>>4)+i, halves 8..15 -> +16.
__device__ __forceinline__ v16h lds_load_b_tile(const _Float16* p) {
    h8 lo = *(const h8*)p;
    h8 hi = *(const h8*)(p + 16);
    v16h r;
#pragma unroll
    for (int i = 0; i < 8; ++i) { r[i] = lo[i]; r[i + 8] = hi[i]; }
    return r;
}

// A fragment per ISA table: halves 0..7 -> K = base+8*khi+0..7, halves 8..15 -> +16
__device__ __forceinline__ v16h lds_load_a_tile(const _Float16* p) {
    h8 lo = *(const h8*)p;
    h8 hi = *(const h8*)(p + 16);
    v16h r;
#pragma unroll
    for (int i = 0; i < 8; ++i) { r[i] = lo[i]; r[i + 8] = hi[i]; }
    return r;
}

__device__ __forceinline__ v8f wmma16(v16h a, v16h b, v8f c) {
    return __builtin_amdgcn_wmma_f32_16x16x32_f16(false, a, false, b, (short)0, c,
                                                  false, false);
}

// ---------------------------------------------------------------------------
// Tensor Data Mover: flat 1-D dword copy global -> LDS.
// D# per CDNA5 ISA ch.8: group0 = {count/flags, lds_addr, global_addr, type=2},
// group1 = {data_size=4B, tensor_dim0=ndw, tensor_dim1=1, tile_dim0=ndw,
//           tile_dim1=1, stride0=ndw}; groups 2/3 zero (<=2D tensor).
// ---------------------------------------------------------------------------
__device__ __forceinline__ void tdm_load_1d(unsigned lds_addr, const void* gptr,
                                            unsigned ndw /* dwords, <= 65535 */) {
    unsigned long long ga = (unsigned long long)(size_t)gptr;
    u32x4 g0;
    g0[0] = 1u;                                         // count=1, user mode
    g0[1] = lds_addr;                                   // LDS byte address
    g0[2] = (unsigned)(ga & 0xffffffffu);               // global_addr[31:0]
    g0[3] = (unsigned)((ga >> 32) & 0x1ffffffu)         // global_addr[56:32]
          | (2u << 30);                                 // type = 2 ("image")
    i32x8 g1;
    g1[0] = (int)(2u << 16);                            // data_size = 4B
    g1[1] = (int)((ndw & 0xffffu) << 16);               // tensor_dim0[15:0]
    g1[2] = (int)((ndw >> 16) | (1u << 16));            // tensor_dim0[31:16] | tensor_dim1=1
    g1[3] = (int)(ndw << 16);                           // tile_dim0 = ndw
    g1[4] = 1;                                          // tile_dim1 = 1 (tile_dim2 = 0)
    g1[5] = (int)ndw;                                   // tensor_dim0_stride[31:0]
    g1[6] = 0;
    g1[7] = 0;
    i32x4 gz;
    gz[0] = 0; gz[1] = 0; gz[2] = 0; gz[3] = 0;
#if defined(__clang_major__) && (__clang_major__ >= 23)
    i32x8 gz8;
#pragma unroll
    for (int i = 0; i < 8; ++i) gz8[i] = 0;
    __builtin_amdgcn_tensor_load_to_lds(g0, g1, gz, gz, gz8, 0);
#else
    __builtin_amdgcn_tensor_load_to_lds(g0, g1, gz, gz, 0);
#endif
}

__device__ __forceinline__ unsigned lds_addr_of(const void* p) {
    // generic pointer to LDS: low 32 bits are the LDS byte offset (ISA 10.2)
    return (unsigned)(size_t)p;
}

// ---------------------------------------------------------------------------
__global__ void zero_f32(float* __restrict__ p, int n) {
    int i = blockIdx.x * blockDim.x + threadIdx.x;
    if (i < n) p[i] = 0.f;
}

// One-time f32 -> f16 transposed/padded weight images (TDM source, LDS layout)
__global__ void prep_weights(const float* __restrict__ rW1,
                             const float* __restrict__ rW2,
                             const float* __restrict__ dW0,
                             const float* __restrict__ dW1,
                             const float* __restrict__ dW2,
                             _Float16* __restrict__ img)
{
    int i = blockIdx.x * blockDim.x + threadIdx.x;
    if (i >= IMG_TOTAL) return;
    float val = 0.f;
    if (i < IMG_W2_OFF) {                       // rW1^T: c*40 + k
        int c = i / W1STRIDE, k = i % W1STRIDE;
        if (k < NRAD) val = rW1[k * HIDDEN + c];
    } else if (i < IMG_K1_HALF) {               // rW2^T: c*136 + k
        int j = i - IMG_W2_OFF;
        int c = j / WSTRIDE, k = j % WSTRIDE;
        if (k < HIDDEN) val = rW2[k * HIDDEN + c];
    } else {                                    // dWl^T
        int j = i - IMG_K1_HALF;
        int l = j / 17408; j %= 17408;
        int c = j / WSTRIDE, k = j % WSTRIDE;
        const float* W = (l == 0) ? dW0 : (l == 1) ? dW1 : dW2;
        if (k < HIDDEN) val = W[k * HIDDEN + c];
    }
    img[i] = (_Float16)val;
}

// ---------------------------------------------------------------------------
// Kernel 1: fused radial MLP (16->128 SiLU 128->128) + gather(h[src]) * W
//           + scatter-add into xacc[dst].  One wave = 16 edges.
// ---------------------------------------------------------------------------
__global__ __launch_bounds__(128)
void radial_gather_scatter(const float* __restrict__ hfeat,
                           const float* __restrict__ rbas,
                           const int*   __restrict__ srcIdx,
                           const int*   __restrict__ dstIdx,
                           const float* __restrict__ rb1,
                           const float* __restrict__ rb2,
                           const _Float16* __restrict__ wimg,
                           float* __restrict__ xacc)
{
    __shared__ alignas(16) _Float16 swts[IMG_K1_HALF];      // W1^T then W2^T
    __shared__ alignas(16) _Float16 stg[4][16 * WSTRIDE];   // per-wave staging

    const int tid = threadIdx.x;
    if (tid < 32) {   // wave 0: single TDM DMA of both weight images into LDS
        tdm_load_1d(lds_addr_of(swts), wimg, IMG_K1_HALF / 2);
        __builtin_amdgcn_s_wait_tensorcnt(0);
    }
    __syncthreads();

    const _Float16* sW1t = swts;
    const _Float16* sW2t = swts + IMG_W2_OFF;

    const int wave = tid >> 5;
    const int lane = tid & 31;
    const int m    = lane & 15;
    const int khi  = lane >> 4;
    const int e0   = blockIdx.x * 64 + wave * 16;
    _Float16* st = &stg[wave][0];

    // ---- A1: radial_basis tile [16 x 16], K padded to 32 with zeros
    v16h a1;
    {
        const float* p = rbas + (size_t)(e0 + m) * NRAD + khi * 8;
        f4 r0 = *(const f4*)p;
        f4 r1 = *(const f4*)(p + 4);
#pragma unroll
        for (int i = 0; i < 4; ++i) {
            a1[i]      = (_Float16)r0[i];
            a1[i + 4]  = (_Float16)r1[i];
            a1[i + 8]  = (_Float16)0.f;
            a1[i + 12] = (_Float16)0.f;
        }
    }

    // ---- GEMM1: [16x32(pad)] x [32x128] -> bias -> SiLU -> f16 staging
#pragma unroll
    for (int n = 0; n < 8; ++n) {
        const int colg = n * 16 + m;
        v16h b = lds_load_b_tile(sW1t + colg * W1STRIDE + khi * 8);
        v8f c = {};
        c = wmma16(a1, b, c);
        const float bias = rb1[colg];
#pragma unroll
        for (int v = 0; v < 8; ++v)
            st[(v + 8 * khi) * WSTRIDE + colg] = (_Float16)silu_f(c[v] + bias);
    }
    asm volatile("" ::: "memory");   // same-wave DS ops are processed in order

    v16h a2[4];
#pragma unroll
    for (int kc = 0; kc < 4; ++kc)
        a2[kc] = lds_load_a_tile(st + m * WSTRIDE + kc * 32 + khi * 8);

    int srcs[8], dsts[8];
#pragma unroll
    for (int v = 0; v < 8; ++v) {
        const int e = e0 + v + 8 * khi;
        srcs[v] = srcIdx[e];
        dsts[v] = dstIdx[e];
    }

    // ---- GEMM2: [16x128] x [128x128] -> +bias -> * h[src] -> atomic scatter
#pragma unroll 2
    for (int n = 0; n < 8; ++n) {
        const int colg = n * 16 + m;
        v8f c = {};
#pragma unroll
        for (int kc = 0; kc < 4; ++kc) {
            v16h b = lds_load_b_tile(sW2t + colg * WSTRIDE + kc * 32 + khi * 8);
            c = wmma16(a2[kc], b, c);
        }
        const float bias = rb2[colg];
#pragma unroll
        for (int v = 0; v < 8; ++v) {
            const float w  = c[v] + bias;
            const float hv = hfeat[(size_t)srcs[v] * HIDDEN + colg];
            atomicAdd(xacc + (size_t)dsts[v] * HIDDEN + colg, w * hv);
        }
    }
}

// ---------------------------------------------------------------------------
// Kernel 2: fused dense x3 (SiLU) + output projection.  One wave = 16 rows.
// Per-layer weights arrive via TDM DMA; activations stay on-chip (per-wave LDS restaging).
// ---------------------------------------------------------------------------
__global__ __launch_bounds__(128)
void dense_layers_out(const float* __restrict__ xin,
                      const float* __restrict__ db0,
                      const float* __restrict__ db1,
                      const float* __restrict__ db2,
                      const float* __restrict__ oW,  const float* __restrict__ ob,
                      const _Float16* __restrict__ wimg,
                      float* __restrict__ out)
{
    __shared__ alignas(16) _Float16 sWt[128 * WSTRIDE];     // current layer weight^T
    __shared__ alignas(16) _Float16 stg[4][16 * WSTRIDE];
    __shared__ float soW[HIDDEN];

    const int tid  = threadIdx.x;
    const int wave = tid >> 5;
    const int lane = tid & 31;
    const int m    = lane & 15;
    const int khi  = lane >> 4;
    const int rowsBase = blockIdx.x * 64 + wave * 16;
    const int row  = rowsBase + m;
    _Float16* st = &stg[wave][0];
    const unsigned sWt_lds = lds_addr_of(sWt);

    soW[tid] = oW[tid];

    // load x rows as f16 A fragments (guarded; tail waves contribute zeros)
    v16h a[4];
#pragma unroll
    for (int kc = 0; kc < 4; ++kc) {
        v16h t;
#pragma unroll
        for (int i = 0; i < 16; ++i) t[i] = (_Float16)0.f;
        if (row < NNODES) {
            const float* p = xin + (size_t)row * HIDDEN + kc * 32 + khi * 8;
            f4 q0 = *(const f4*)p;
            f4 q1 = *(const f4*)(p + 4);
            f4 q2 = *(const f4*)(p + 16);
            f4 q3 = *(const f4*)(p + 20);
#pragma unroll
            for (int i = 0; i < 4; ++i) {
                t[i]      = (_Float16)q0[i];
                t[i + 4]  = (_Float16)q1[i];
                t[i + 8]  = (_Float16)q2[i];
                t[i + 12] = (_Float16)q3[i];
            }
        }
        a[kc] = t;
    }

    float part[8];
#pragma unroll
    for (int v = 0; v < 8; ++v) part[v] = 0.f;

    for (int l = 0; l < 3; ++l) {
        const float* bl = (l == 0) ? db0 : (l == 1) ? db1 : db2;

        __syncthreads();   // all waves done reading previous layer's sWt
        if (tid < 32) {    // wave 0: DMA this layer's f16 weight image into LDS
            tdm_load_1d(sWt_lds, wimg + IMG_DENSE(l), 17408 / 2);
            __builtin_amdgcn_s_wait_tensorcnt(0);
        }
        __syncthreads();

#pragma unroll 2
        for (int n = 0; n < 8; ++n) {
            const int colg = n * 16 + m;
            v8f c = {};
#pragma unroll
            for (int kc = 0; kc < 4; ++kc) {
                v16h b = lds_load_b_tile(sWt + colg * WSTRIDE + kc * 32 + khi * 8);
                c = wmma16(a[kc], b, c);
            }
            const float bias = bl[colg];
            if (l < 2) {
#pragma unroll
                for (int v = 0; v < 8; ++v)
                    st[(v + 8 * khi) * WSTRIDE + colg] = (_Float16)silu_f(c[v] + bias);
            } else {
                // fold the [128 x 1] output projection into layer-2 epilogue
                const float ow = soW[colg];
#pragma unroll
                for (int v = 0; v < 8; ++v)
                    part[v] += silu_f(c[v] + bias) * ow;
            }
        }
        if (l < 2) {
            asm volatile("" ::: "memory");
#pragma unroll
            for (int kc = 0; kc < 4; ++kc)
                a[kc] = lds_load_a_tile(st + m * WSTRIDE + kc * 32 + khi * 8);
            asm volatile("" ::: "memory");
        }
    }

    // cross-lane reduction over the 16 columns held by each 16-lane half
#pragma unroll
    for (int v = 0; v < 8; ++v) {
        float s = part[v];
        s += __shfl_xor(s, 1, 32);
        s += __shfl_xor(s, 2, 32);
        s += __shfl_xor(s, 4, 32);
        s += __shfl_xor(s, 8, 32);
        part[v] = s;
    }
    if (m == 0) {
        const float ob0 = ob[0];
#pragma unroll
        for (int v = 0; v < 8; ++v) {
            const int r = rowsBase + v + 8 * khi;
            if (r < NNODES) out[r] = part[v] + ob0;
        }
    }
}

// ---------------------------------------------------------------------------
extern "C" void kernel_launch(void* const* d_in, const int* in_sizes, int n_in,
                              void* d_out, int out_size, void* d_ws, size_t ws_size,
                              hipStream_t stream)
{
    (void)in_sizes; (void)n_in; (void)out_size; (void)ws_size;

    const float* h    = (const float*)d_in[0];
    const float* rbas = (const float*)d_in[1];
    const int*   eidx = (const int*)  d_in[2];
    const float* rW1  = (const float*)d_in[3];
    const float* rb1  = (const float*)d_in[4];
    const float* rW2  = (const float*)d_in[5];
    const float* rb2  = (const float*)d_in[6];
    const float* dW0  = (const float*)d_in[7];
    const float* db0  = (const float*)d_in[8];
    const float* dW1  = (const float*)d_in[9];
    const float* db1  = (const float*)d_in[10];
    const float* dW2  = (const float*)d_in[11];
    const float* db2  = (const float*)d_in[12];
    const float* oW   = (const float*)d_in[13];
    const float* ob   = (const float*)d_in[14];

    const int* src = eidx;
    const int* dst = eidx + NEDGES;

    float*     xacc = (float*)d_ws;                       // [NNODES, HIDDEN] f32
    _Float16*  wimg = (_Float16*)(xacc + NNODES * HIDDEN);// f16 weight images
    float*     out  = (float*)d_out;                      // [NNODES] f32

    const int nacc = NNODES * HIDDEN;
    zero_f32<<<(nacc + 255) / 256, 256, 0, stream>>>(xacc, nacc);
    prep_weights<<<(IMG_TOTAL + 255) / 256, 256, 0, stream>>>(
        rW1, rW2, dW0, dW1, dW2, wimg);

    radial_gather_scatter<<<NEDGES / 64, 128, 0, stream>>>(
        h, rbas, src, dst, rb1, rb2, wimg, xacc);

    dense_layers_out<<<(NNODES + 63) / 64, 128, 0, stream>>>(
        xacc, db0, db1, db2, oW, ob, wimg, out);
}
